// DMICE_21517786153523
// MI455X (gfx1250) — matrix operand order repaired and checked by reading, not compiled, not run on hardware
//
#include <hip/hip_runtime.h>

// Problem constants (match reference)
#define B_      256
#define L_      16384
#define IGNORE_ 3
// Launch geometry
#define SEG            8                       // L split into 8 segments -> 2048 blocks
#define TPB            256                     // 8 waves (wave32)
#define WAVES          (TPB / 32)
#define TOK_PER_BLOCK  (L_ / SEG)              // 2048
#define TOK_PER_WAVE   (TOK_PER_BLOCK / WAVES) // 256
#define CHUNKS         (TOK_PER_WAVE / 32)     // 8 WMMA steps per wave

typedef __attribute__((ext_vector_type(16))) _Float16 v16h;
typedef __attribute__((ext_vector_type(8)))  _Float16 v8h;
typedef __attribute__((ext_vector_type(8)))  float    v8f;

// ---------------------------------------------------------------- init ws
// ws layout (floats): [0..2303] mat accum (256 samples x 9), [2304] ce_sum,
// [2305] valid_cnt, then int jarr[256] at float offset 2306.
__global__ __launch_bounds__(256) void k_init(float* __restrict__ ws) {
  int i = blockIdx.x * 256 + threadIdx.x;
  if (i < 2306) ws[i] = 0.0f;
}

// ------------------------------------------------------- first pad index j
__global__ __launch_bounds__(256) void k_findj(const int* __restrict__ labels,
                                               int* __restrict__ jarr) {
  const int b = blockIdx.x;
  const int4* lb = (const int4*)(labels + (size_t)b * L_);
  int best = L_;
  for (int i = threadIdx.x; i < L_ / 4; i += 256) {
    int4 v = lb[i];
    int base = i * 4;
    if (v.x == IGNORE_) best = min(best, base + 0);
    if (v.y == IGNORE_) best = min(best, base + 1);
    if (v.z == IGNORE_) best = min(best, base + 2);
    if (v.w == IGNORE_) best = min(best, base + 3);
  }
  __shared__ int red[256];
  red[threadIdx.x] = best;
  __syncthreads();
  for (int s = 128; s > 0; s >>= 1) {
    if (threadIdx.x < s) red[threadIdx.x] = min(red[threadIdx.x], red[threadIdx.x + s]);
    __syncthreads();
  }
  if (threadIdx.x == 0) jarr[b] = (red[0] == L_) ? (L_ - 1) : red[0];
}

// ---------------------------------------------------------------- main pass
// Fused: CE partial sums + WMMA accumulation of matT = probs(16x32) x onehot(32x16).
__global__ __launch_bounds__(TPB) void k_main(const float* __restrict__ pred,
                                              const int*   __restrict__ labels,
                                              const int*   __restrict__ jarr,
                                              float* __restrict__ wsmat,
                                              float* __restrict__ ce_cnt) {
  const int b    = blockIdx.y;
  const int seg  = blockIdx.x;
  const int tid  = threadIdx.x;
  const int w    = tid >> 5;
  const int lane = tid & 31;
  const int jb   = jarr[b];

  // probs staging, CLASS-major: [wave][class0..2 + zero row][token-in-chunk]
  // Row stride 64B, 16B-aligned -> each A half-row is one ds_load_b128.
  __shared__ __align__(16) _Float16 sp[WAVES][4][32];

  const float* p0p = pred + ((size_t)b * 4 + 0) * L_;
  const float* p1p = pred + ((size_t)b * 4 + 1) * L_;
  const float* p2p = pred + ((size_t)b * 4 + 2) * L_;
  const float* p3p = pred + ((size_t)b * 4 + 3) * L_;
  const int*   lbp = labels + (size_t)b * L_;

  // A-operand layout (16-bit A 16x32, ISA 7.12.2): lane m<16 holds row m,
  // K = {0..7} in elems 0..7 and {16..23} in elems 8..15; lane m+16 holds
  // K = {8..15} and {24..31}.
  const int row = lane & 15;
  const int rr  = (row < 3) ? row : 3;          // rows 3..15 read the zero row
  const int klo = (lane & 16) ? 8 : 0;
  const int khi = klo + 16;
  // B-operand layout (16-bit B 32x16), matching A: column = lane%16,
  // lanes 0-15 hold K={0..7,16..23}, lanes 16-31 hold K={8..15,24..31}.
  const int col = row;
  const int kb  = (lane & 16) ? 16 : 0;
  // Loop-invariant per-lane masks for branchless onehot:
  //   (lab == col)  <=>  ((bb0 ^ inv0) & (bb1 ^ inv1)),  col<3 gate via czm.
  const unsigned inv0 = (col & 1) ? 0u : ~0u;
  const unsigned inv1 = (col & 2) ? 0u : ~0u;
  const unsigned czm  = (col < 3) ? ~0u : 0u;

  sp[w][3][lane] = (_Float16)0.0f;              // zero row, written once
  __builtin_amdgcn_wave_barrier();

  float ce_acc = 0.0f, cnt_acc = 0.0f;
  v8f c = {};                                    // matT accumulator (f32)

  const int t0 = seg * TOK_PER_BLOCK + w * TOK_PER_WAVE;
#pragma unroll
  for (int it = 0; it < CHUNKS; ++it) {
    const int token = t0 + it * 32 + lane;       // one token per lane, K=lane
    const int lab = lbp[token];
    const float x0 = p0p[token], x1 = p1p[token], x2 = p2p[token], x3 = p3p[token];

    // log-softmax over 4 classes (CE), softmax over first 3 reusing the exps
    const float m4 = fmaxf(fmaxf(x0, x1), fmaxf(x2, x3));
    const float e0 = __expf(x0 - m4), e1 = __expf(x1 - m4);
    const float e2 = __expf(x2 - m4), e3 = __expf(x3 - m4);
    const float lse = m4 + __logf(e0 + e1 + e2 + e3);
    // branchless pick (monotone cndmask chain, keeps EXEC uniform)
    float picked = x0;
    picked = (lab == 1) ? x1 : picked;
    picked = (lab == 2) ? x2 : picked;
    picked = (lab == 3) ? x3 : picked;
    const float isv = (lab != IGNORE_) ? 1.0f : 0.0f;
    ce_acc  = fmaf(lse - picked, isv, ce_acc);
    cnt_acc += isv;

    // softmax3 shares the m4 shift; fast rcp (v_rcp_f32), no IEEE divide chain
    const float inv3 = __builtin_amdgcn_rcpf(e0 + e1 + e2);
    sp[w][0][lane] = (_Float16)(e0 * inv3);
    sp[w][1][lane] = (_Float16)(e1 * inv3);
    sp[w][2][lane] = (_Float16)(e2 * inv3);

    // labels(2 bits) + prefix mask broadcast to all lanes via wave32 ballots
    const unsigned bb0 = __builtin_amdgcn_ballot_w32((lab & 1) != 0);
    const unsigned bb1 = __builtin_amdgcn_ballot_w32((lab & 2) != 0);
    const unsigned vm  = __builtin_amdgcn_ballot_w32(token < jb);
    // branchless: onehot(lab_k == col) & (k < j) & (col < 3), no EXEC games
    const unsigned sel = (bb0 ^ inv0) & (bb1 ^ inv1) & vm & czm;

    __builtin_amdgcn_wave_barrier();             // keep DS write->read ordering

    // A = probs rows: two contiguous 16B LDS reads per lane (ds_load_b128)
    const v8h* rowp = (const v8h*)(&sp[w][rr][0]);
    const v8h alo = rowp[klo >> 3];
    const v8h ahi = rowp[khi >> 3];
    const v16h A = __builtin_shufflevector(alo, ahi,
                                           0, 1, 2, 3, 4, 5, 6, 7,
                                           8, 9, 10, 11, 12, 13, 14, 15);
    v16h Bv;
#pragma unroll
    for (int e = 0; e < 16; ++e) {               // B = masked onehot (0/1 f16)
      Bv[e] = ((sel >> (klo + e + ((e < 8) ? 0 : 8))) & 1u) ? (_Float16)1.0f : (_Float16)0.0f;
    }
    // D = A x B + C  ->  matT[m][n] += sum_k probs_m(k) * onehot(lab_k==n)
    c = __builtin_amdgcn_wmma_f32_16x16x32_f16(false, A, false, Bv,
                                               (short)0, c, false, false);
  }

  // CE wave reduction + global accumulate
  for (int off = 16; off > 0; off >>= 1) {
    ce_acc  += __shfl_xor(ce_acc,  off, 32);
    cnt_acc += __shfl_xor(cnt_acc, off, 32);
  }
  if (lane == 0) {
    atomicAdd(&ce_cnt[0], ce_acc);
    atomicAdd(&ce_cnt[1], cnt_acc);
  }
  // D layout: VGPR r, lane n<16 -> (M=r, N=n). Lane i (i<3) holds matT[0..2][i]
  // in c[0..2], i.e. mat[i][0..2].
  if (lane < 3) {
    atomicAdd(&wsmat[b * 9 + lane * 3 + 0], c[0]);
    atomicAdd(&wsmat[b * 9 + lane * 3 + 1], c[1]);
    atomicAdd(&wsmat[b * 9 + lane * 3 + 2], c[2]);
  }
}

// -------------------------------------------------------------- finalize
__global__ __launch_bounds__(256) void k_final(const float* __restrict__ wsmat,
                                               const int*   __restrict__ jarr,
                                               const float* __restrict__ ce_cnt,
                                               float* __restrict__ out) {
  const int b = threadIdx.x;
  const float inv = 1.0f / (float)jarr[b];
  float m[9];
#pragma unroll
  for (int i = 0; i < 9; ++i) m[i] = wsmat[b * 9 + i] * inv;
  const float det = m[0] * (m[4] * m[8] - m[5] * m[7])
                  - m[1] * (m[3] * m[8] - m[5] * m[6])
                  + m[2] * (m[3] * m[7] - m[4] * m[6]);
  const float ad  = fabsf(det) + 1e-3f;
  const float dmi = (det < 0.0f) ? __logf(ad) : -__logf(ad);

  __shared__ float red[256];
  red[b] = dmi;
  __syncthreads();
  for (int s = 128; s > 0; s >>= 1) {
    if (b < s) red[b] += red[b + s];
    __syncthreads();
  }
  if (b == 0) out[0] = 0.1f * (red[0] / (float)B_) + ce_cnt[0] / ce_cnt[1];
}

// ---------------------------------------------------------------- launcher
extern "C" void kernel_launch(void* const* d_in, const int* in_sizes, int n_in,
                              void* d_out, int out_size, void* d_ws, size_t ws_size,
                              hipStream_t stream) {
  const float* pred   = (const float*)d_in[0];
  const int*   labels = (const int*)d_in[1];

  float* ws     = (float*)d_ws;
  float* wsmat  = ws;                 // 2304 floats
  float* ce_cnt = ws + 2304;          // 2 floats
  int*   jarr   = (int*)(ws + 2306);  // 256 ints

  k_init<<<10, 256, 0, stream>>>(ws);
  k_findj<<<B_, 256, 0, stream>>>(labels, jarr);
  dim3 grid(SEG, B_);
  k_main<<<grid, TPB, 0, stream>>>(pred, labels, jarr, wsmat, ce_cnt);
  k_final<<<1, 256, 0, stream>>>(wsmat, jarr, ce_cnt, (float*)d_out);
}
